// TrainiumBasedLinearAttention_40948218200458
// MI455X (gfx1250) — compile-verified, matrix operand order checked by reading, not run on hardware
//
#include <hip/hip_runtime.h>
#include <hip/hip_bf16.h>

// ---------------------------------------------------------------------------
// BASED linear attention (2nd-order Taylor softmax approx) for MI455X/gfx1250.
// b=2, t=2048, dim=1024, heads=16, d=64. Everything routed through
// v_wmma_f32_16x16x32_bf16; working set (~48MB bf16) stays L2-resident.
// ---------------------------------------------------------------------------

#define B_      2
#define SEQ_    2048
#define DIM_    1024
#define HEADS_  16
#define DHEAD_  64
#define BT_     (B_ * SEQ_)         // 4096
#define EPS_    1e-6f

typedef __attribute__((ext_vector_type(8)))  __bf16 v8bf;
typedef __attribute__((ext_vector_type(16))) __bf16 v16bf;
typedef __attribute__((ext_vector_type(8)))  float  v8f;

// ---- helpers --------------------------------------------------------------

__device__ __forceinline__ unsigned short f32_to_bf16(float f) {
  unsigned int u = __float_as_uint(f);
  unsigned int lsb = (u >> 16) & 1u;
  u += 0x7fffu + lsb;                 // round-to-nearest-even
  return (unsigned short)(u >> 16);
}

// Load one 16x32 bf16 WMMA operand fragment from a row-major matrix.
// CDNA5 16-bit A layout: lanes 0-15 hold row (i0+lane), K = {0..7, 16..23};
// lanes 16-31 hold row (i0+lane-16), K = {8..15, 24..31}. The B operand
// (column n fixed per lane, K along elements) uses the same loader on the
// row-major "other" matrix (rows of W == columns of W^T).
__device__ __forceinline__ v16bf load_frag_rm(const unsigned short* p, int ld,
                                              int i0, int k0, int lane) {
  const int hi16 = lane >> 4;         // 0 or 1
  const int l    = lane & 15;
  const unsigned short* base = p + (size_t)(i0 + l) * (size_t)ld + k0 + hi16 * 8;
  v8bf lo = *reinterpret_cast<const v8bf*>(base);        // K 0..7   (+hi16*8)
  v8bf hi = *reinterpret_cast<const v8bf*>(base + 16);   // K 16..23 (+hi16*8)
  v16bf f;
#pragma unroll
  for (int i = 0; i < 8; ++i) { f[i] = lo[i]; f[i + 8] = hi[i]; }
  return f;
}

__device__ __forceinline__ v8f wmma_bf16(v16bf a, v16bf b, v8f c) {
  // D = A(16x32) * B(32x16) + C, f32 accumulate
  return __builtin_amdgcn_wmma_f32_16x16x32_bf16(
      false, a, false, b, (short)0, c, false, false);
}

// ---- fp32 -> bf16 convert -------------------------------------------------

__global__ void cvt_f32_bf16_kernel(const float* __restrict__ in,
                                    unsigned short* __restrict__ out, int n) {
  int i = blockIdx.x * blockDim.x + threadIdx.x;
  if (i < n) out[i] = f32_to_bf16(in[i]);
}

// ---- GEMM: Out = alpha * (A[M,K] @ B[N,K]^T) ------------------------------
// mode 0: Out bf16 row-major [M,N]
// mode 1: Out bf16 per-head transposed vT[b, h, dcol, t]  (for the V tensor)
// mode 2: Out f32 row-major [M,N]                          (final projection)

#define OUT_BF16_RM 0
#define OUT_BF16_VT 1
#define OUT_F32_RM  2

__global__ __launch_bounds__(256)
void gemm_xwT_kernel(const unsigned short* __restrict__ A,
                     const unsigned short* __restrict__ Bm,
                     void* __restrict__ Out,
                     int M, int N, int K, float alpha, int mode) {
  const int lane = threadIdx.x & 31;
  const int w    = threadIdx.x >> 5;               // 8 waves
  const int m0   = blockIdx.y * 128 + (w >> 2) * 64;  // wave: 64 rows
  const int n0   = blockIdx.x * 128 + (w & 3) * 32;   // wave: 32 cols

  v8f acc[4][2];
#pragma unroll
  for (int i = 0; i < 4; ++i)
#pragma unroll
    for (int j = 0; j < 2; ++j)
#pragma unroll
      for (int r = 0; r < 8; ++r) acc[i][j][r] = 0.0f;

  for (int kb = 0; kb < K; kb += 32) {
    v16bf bf[2];
#pragma unroll
    for (int j = 0; j < 2; ++j)
      bf[j] = load_frag_rm(Bm, K, n0 + j * 16, kb, lane);
#pragma unroll
    for (int i = 0; i < 4; ++i) {
      v16bf af = load_frag_rm(A, K, m0 + i * 16, kb, lane);
#pragma unroll
      for (int j = 0; j < 2; ++j)
        acc[i][j] = wmma_bf16(af, bf[j], acc[i][j]);
    }
  }

  // C/D layout: VGPR r -> row M = r + 8*(lane>=16); col N = lane & 15
  const int half8 = (lane >> 4) << 3;
  const int ncol  = lane & 15;
#pragma unroll
  for (int i = 0; i < 4; ++i) {
#pragma unroll
    for (int j = 0; j < 2; ++j) {
#pragma unroll
      for (int r = 0; r < 8; ++r) {
        const int mm = m0 + i * 16 + r + half8;
        const int nn = n0 + j * 16 + ncol;
        const float val = acc[i][j][r] * alpha;
        if (mode == OUT_F32_RM) {
          ((float*)Out)[(size_t)mm * N + nn] = val;
        } else if (mode == OUT_BF16_RM) {
          ((unsigned short*)Out)[(size_t)mm * N + nn] = f32_to_bf16(val);
        } else {  // OUT_BF16_VT: v[b*t + t_loc, h*64+dc] -> vT[b, h, dc, t_loc]
          const int bb = mm >> 11;            // /2048
          const int tl = mm & (SEQ_ - 1);
          const int hh = nn >> 6;             // /64
          const int dc = nn & (DHEAD_ - 1);
          ((unsigned short*)Out)[((size_t)((bb * HEADS_ + hh) * DHEAD_ + dc)) * SEQ_ + tl]
              = f32_to_bf16(val);
        }
      }
    }
  }
}

// ---- BASED causal attention ----------------------------------------------
// grid = (SEQ/64, B*HEADS), 128 threads = 4 waves; each wave owns a 16-query
// WMMA tile. attn = 1 + s + 0.5*s^2 masked causally; o = attn@v; z = rowsum.

__global__ __launch_bounds__(128)
void based_attn_kernel(const unsigned short* __restrict__ Qb,   // [BT, DIM] bf16 (pre-scaled)
                       const unsigned short* __restrict__ Kb,   // [BT, DIM] bf16
                       const unsigned short* __restrict__ VT,   // [B,H,d,t] bf16
                       unsigned short* __restrict__ Ob) {       // [BT, DIM] bf16
  __shared__ unsigned short sA[4][16][40];   // wave-private attn tile, padded rows

  const int lane = threadIdx.x & 31;
  const int w    = threadIdx.x >> 5;                 // 0..3
  const int qblk = blockIdx.x;                       // 0..31
  const int bh   = blockIdx.y;                       // 0..31
  const int b    = bh >> 4;
  const int h    = bh & (HEADS_ - 1);

  const int q0 = qblk * 64 + w * 16;                 // first query row (t index)

  const unsigned short* qp = Qb + (size_t)b * SEQ_ * DIM_ + h * DHEAD_;
  const unsigned short* kp = Kb + (size_t)b * SEQ_ * DIM_ + h * DHEAD_;
  const unsigned short* vp = VT + ((size_t)(b * HEADS_ + h)) * DHEAD_ * SEQ_;

  // q fragments for K-dim d=64 (two 32-wide steps), kept resident
  const v16bf qa0 = load_frag_rm(qp, DIM_, q0, 0, lane);
  const v16bf qa1 = load_frag_rm(qp, DIM_, q0, 32, lane);

  v8f oacc[4];
  v8f zacc;
#pragma unroll
  for (int j = 0; j < 4; ++j)
#pragma unroll
    for (int r = 0; r < 8; ++r) oacc[j][r] = 0.0f;
#pragma unroll
  for (int r = 0; r < 8; ++r) zacc[r] = 0.0f;

  const int half8 = (lane >> 4) << 3;
  const int ncol  = lane & 15;
  const int nkb   = (q0 + 47) >> 5;                  // causal #32-key blocks

  for (int kb2 = 0; kb2 < nkb; ++kb2) {
    const int key0 = kb2 * 32;

    // s = q . k for 16 queries x 32 keys (two 16x16 N-tiles, K=64)
#pragma unroll
    for (int t = 0; t < 2; ++t) {
      v8f s;
#pragma unroll
      for (int r = 0; r < 8; ++r) s[r] = 0.0f;
      v16bf kf0 = load_frag_rm(kp, DIM_, key0 + t * 16, 0, lane);
      v16bf kf1 = load_frag_rm(kp, DIM_, key0 + t * 16, 32, lane);
      s = wmma_bf16(qa0, kf0, s);
      s = wmma_bf16(qa1, kf1, s);

      // attn = 1 + s + 0.5 s^2, causal mask; accumulate z; stage to LDS
#pragma unroll
      for (int r = 0; r < 8; ++r) {
        const int M  = r + half8;
        const int qi = q0 + M;
        const int kj = key0 + t * 16 + ncol;
        const float sv = s[r];
        const float a  = (kj <= qi) ? __builtin_fmaf(0.5f * sv, sv, sv) + 1.0f : 0.0f;
        zacc[r] += a;
        sA[w][M][t * 16 + ncol] = f32_to_bf16(a);
      }
    }

    // Re-read attn as an A-layout fragment (same-wave DS ops stay in order)
    const v16bf af = load_frag_rm(&sA[w][0][0], 40, 0, 0, lane);

    // o(16x64) += attn(16x32) @ v(32x64): 4 N-tiles, K=32 keys
#pragma unroll
    for (int j = 0; j < 4; ++j) {
      v16bf vf = load_frag_rm(vp, SEQ_, j * 16, key0, lane);
      oacc[j] = wmma_bf16(af, vf, oacc[j]);
    }
  }

  // z rowsum: reduce partial sums across the 16-lane half (wave32)
  float zr[8];
#pragma unroll
  for (int r = 0; r < 8; ++r) {
    float z = zacc[r];
    z += __shfl_xor(z, 1, 16);
    z += __shfl_xor(z, 2, 16);
    z += __shfl_xor(z, 4, 16);
    z += __shfl_xor(z, 8, 16);
    zr[r] = z + EPS_;
  }

  // normalize and store o back in [b, t, dim] layout (heads un-transposed)
#pragma unroll
  for (int j = 0; j < 4; ++j) {
#pragma unroll
    for (int r = 0; r < 8; ++r) {
      const int M = r + half8;
      const size_t row = (size_t)b * SEQ_ + q0 + M;
      const size_t col = (size_t)h * DHEAD_ + j * 16 + ncol;
      Ob[row * DIM_ + col] = f32_to_bf16(oacc[j][r] / zr[r]);
    }
  }
}

// ---------------------------------------------------------------------------

extern "C" void kernel_launch(void* const* d_in, const int* in_sizes, int n_in,
                              void* d_out, int out_size, void* d_ws, size_t ws_size,
                              hipStream_t stream) {
  (void)in_sizes; (void)n_in; (void)out_size; (void)ws_size;

  const float* X  = (const float*)d_in[0];   // [2,2048,1024]
  const float* Wq = (const float*)d_in[1];   // [1024,1024]
  const float* Wk = (const float*)d_in[2];
  const float* Wv = (const float*)d_in[3];
  const float* Wo = (const float*)d_in[4];

  // bf16 scratch layout (ushort elements), total = 48 MB
  unsigned short* ws  = (unsigned short*)d_ws;
  unsigned short* Xb  = ws;                          // 4096*1024
  unsigned short* Wqb = Xb  + (size_t)BT_ * DIM_;    // 1024*1024 each
  unsigned short* Wkb = Wqb + (size_t)DIM_ * DIM_;
  unsigned short* Wvb = Wkb + (size_t)DIM_ * DIM_;
  unsigned short* Wob = Wvb + (size_t)DIM_ * DIM_;
  unsigned short* Qb  = Wob + (size_t)DIM_ * DIM_;   // 4096*1024
  unsigned short* Kb  = Qb  + (size_t)BT_ * DIM_;
  unsigned short* VT  = Kb  + (size_t)BT_ * DIM_;    // [B,H,d,t]
  unsigned short* Ob  = VT  + (size_t)BT_ * DIM_;

  // 1) fp32 -> bf16
  {
    const int nX = BT_ * DIM_, nW = DIM_ * DIM_;
    cvt_f32_bf16_kernel<<<(nX + 255) / 256, 256, 0, stream>>>(X, Xb, nX);
    cvt_f32_bf16_kernel<<<(nW + 255) / 256, 256, 0, stream>>>(Wq, Wqb, nW);
    cvt_f32_bf16_kernel<<<(nW + 255) / 256, 256, 0, stream>>>(Wk, Wkb, nW);
    cvt_f32_bf16_kernel<<<(nW + 255) / 256, 256, 0, stream>>>(Wv, Wvb, nW);
    cvt_f32_bf16_kernel<<<(nW + 255) / 256, 256, 0, stream>>>(Wo, Wob, nW);
  }

  // 2) projections: q = 0.125 * X Wq^T ; k = X Wk^T ; v = X Wv^T (stored [b,h,d,t])
  dim3 ggrid(DIM_ / 128, BT_ / 128);      // (8, 32)
  gemm_xwT_kernel<<<ggrid, 256, 0, stream>>>(Xb, Wqb, Qb, BT_, DIM_, DIM_,
                                             0.125f /* d^-0.5 */, OUT_BF16_RM);
  gemm_xwT_kernel<<<ggrid, 256, 0, stream>>>(Xb, Wkb, Kb, BT_, DIM_, DIM_,
                                             1.0f, OUT_BF16_RM);
  gemm_xwT_kernel<<<ggrid, 256, 0, stream>>>(Xb, Wvb, VT, BT_, DIM_, DIM_,
                                             1.0f, OUT_BF16_VT);

  // 3) causal BASED attention -> Ob [b,t,dim] bf16
  based_attn_kernel<<<dim3(SEQ_ / 64, B_ * HEADS_), 128, 0, stream>>>(Qb, Kb, VT, Ob);

  // 4) y = o @ Wo^T -> fp32 d_out
  gemm_xwT_kernel<<<ggrid, 256, 0, stream>>>(Ob, Wob, d_out, BT_, DIM_, DIM_,
                                             1.0f, OUT_F32_RM);
}
